// Equalize_13340168422043
// MI455X (gfx1250) — matrix operand order repaired and checked by reading, not compiled, not run on hardware
//
#include <hip/hip_runtime.h>

#define N_BINS 256
#define INV255 (1.0f / 255.0f)
#define NEG_HALF_INV_TAU2 (-5000.0f)   // -1/(2*0.01^2)
#define EPSV 1e-10f
#define NBATCH 4

typedef __attribute__((ext_vector_type(16))) _Float16 v16h;
typedef __attribute__((ext_vector_type(8)))  float    v8f;

// ---------------- kernel 0: zero the histogram accumulator ----------------
__global__ void zero_ws_kernel(float* ws) {
    ws[threadIdx.x] = 0.0f;            // 1024 threads = 4 batches * 256 bins
}

// ---------------- kernel 1: windowed soft histogram -----------------------
// Beyond ~|Δbin| > 34, exp(-5000*d*d) underflows to exactly 0.0f, so a
// +/-40 bin window produces bit-identical partial sums to the dense version.
__global__ void hist_kernel(const float* __restrict__ x,
                            float* __restrict__ ghist,
                            int pixPerBatch, int pixPerBlock) {
    __shared__ float lhist[N_BINS];
    const int t = threadIdx.x;
    lhist[t] = 0.0f;
    __syncthreads();

    const int blocksPerBatch = pixPerBatch / pixPerBlock;
    const int b  = blockIdx.x / blocksPerBatch;
    const int p0 = (blockIdx.x % blocksPerBatch) * pixPerBlock;
    const float* xb = x + (size_t)b * pixPerBatch;

    for (int p = p0 + t; p < p0 + pixPerBlock; p += blockDim.x) {
        const float xv = xb[p];
        int c  = (int)(xv * 255.0f + 0.5f);
        int jl = c - 40; if (jl < 0)   jl = 0;
        int jh = c + 40; if (jh > 255) jh = 255;
        for (int j = jl; j <= jh; ++j) {
            const float d = xv - (float)j * INV255;
            atomicAdd(&lhist[j], __expf(NEG_HALF_INV_TAU2 * d * d)); // ds_add_f32
        }
    }
    __syncthreads();
    atomicAdd(&ghist[b * N_BINS + t], lhist[t]);   // global_atomic_add_f32
}

// ---------------- kernel 2: normalize + scan -> normalized CDF ------------
__global__ void cdf_kernel(const float* __restrict__ ghist,
                           float* __restrict__ cdfn) {
    __shared__ float sh[N_BINS];
    const int t = threadIdx.x;                     // 256 threads
    for (int b = 0; b < NBATCH; ++b) {
        const float h = ghist[b * N_BINS + t];
        // total mass (tree reduction)
        sh[t] = h;
        __syncthreads();
        for (int s = N_BINS / 2; s > 0; s >>= 1) {
            if (t < s) sh[t] += sh[t + s];
            __syncthreads();
        }
        const float total = sh[0];
        __syncthreads();

        // inclusive scan of normalized histogram (Hillis-Steele)
        sh[t] = h / (total + EPSV);
        __syncthreads();
        for (int off = 1; off < N_BINS; off <<= 1) {
            const float add = (t >= off) ? sh[t - off] : 0.0f;
            __syncthreads();
            sh[t] += add;
            __syncthreads();
        }
        const float cdf  = sh[t];
        const float cmin = sh[0];
        cdfn[b * N_BINS + t] = (cdf - cmin) / (1.0f - cmin + EPSV);
        __syncthreads();
    }
}

// ---------------- kernel 3: per-pixel equalization via WMMA ---------------
// Per wave: 16-pixel tile. D(16x16,f32) = A(16x256,f16 weights) x B(256x16,f16)
// where B col0 = cdfn, col1 = 1.0.  => D[r][0] = sum(w*cdfn), D[r][1] = sum(w).
__global__ void equalize_wmma_kernel(const float* __restrict__ x,
                                     const float* __restrict__ cdfn,
                                     float* __restrict__ out,
                                     int pixPerBatch) {
    __shared__ __align__(16) float scdf[N_BINS];
    const int t    = threadIdx.x;                  // 256 threads = 8 waves
    const int lane = t & 31;
    const int wave = t >> 5;

    const int tilesPerBatch  = pixPerBatch / 16;
    const int blocksPerBatch = tilesPerBatch / 8;
    const int b    = blockIdx.x / blocksPerBatch;
    const int tile = (blockIdx.x % blocksPerBatch) * 8 + wave;

    if (t < N_BINS) scdf[t] = cdfn[b * N_BINS + t];
    __syncthreads();

    const int m  = lane & 15;                      // A-row / B-column index
    const int hi = lane >> 4;                      // half-wave select
    const size_t base = (size_t)b * pixPerBatch + (size_t)tile * 16;
    const float xv = x[base + m];                  // pixel value for this A row

    // Branchless B-column selectors: col0 gets cdfn, col1 gets 1.0, rest 0.
    const float sel0 = (m == 0) ? 1.0f : 0.0f;     // v_cndmask, no branch
    const float sel1 = (m == 1) ? 1.0f : 0.0f;

    v8f c = {};
    #pragma unroll
    for (int step = 0; step < 8; ++step) {
        const int kbase = step * 32;

        // Unconditional LDS vector loads of this half-wave's 16 cdfn values
        // (4 x ds_load_b128), then blend arithmetically -> no EXEC branching.
        const float4* cp = (const float4*)&scdf[kbase + hi * 16];
        float cv[16];
        #pragma unroll
        for (int q = 0; q < 4; ++q) {
            const float4 f = cp[q];
            cv[q * 4 + 0] = f.x; cv[q * 4 + 1] = f.y;
            cv[q * 4 + 2] = f.z; cv[q * 4 + 3] = f.w;
        }

        v16h a, bm;
        #pragma unroll
        for (int e = 0; e < 16; ++e) {
            // A 16x32 f16, contiguous k order used here: lanes<16 hold K=0..15, lanes>=16 hold K=16..31
            const int ka = hi * 16 + e;
            const float d = xv - (float)(kbase + ka) * INV255;
            a[e] = (_Float16)__expf(NEG_HALF_INV_TAU2 * d * d);
            // B 32x16 f16, same contiguous k order as A: N = lane&15; lanes<16 hold K=0..15, lanes>=16 hold K=16..31
            bm[e] = (_Float16)__builtin_fmaf(sel0, cv[e], sel1);   // v_fma_f32
        }
        c = __builtin_amdgcn_wmma_f32_16x16x32_f16(
                /*neg_a=*/false, a, /*neg_b=*/false, bm,
                /*c_mod=*/(short)0, c, /*reuse_a=*/false, /*reuse_b=*/false);
    }

    // C/D 16x16 f32 layout: VGPR g holds M=g (lanes 0-15, N=lane) and M=g+8
    // (lanes 16-31, N=lane-16).  So lanes {0,16} hold numerators, {1,17} denominators.
    #pragma unroll
    for (int g = 0; g < 8; ++g) {
        const float den = __shfl_xor(c[g], 1, 32);
        if (m == 0) {
            out[base + hi * 8 + g] = c[g] / (den + EPSV);
        }
    }
}

// --------------------------------------------------------------------------
extern "C" void kernel_launch(void* const* d_in, const int* in_sizes, int n_in,
                              void* d_out, int out_size, void* d_ws, size_t ws_size,
                              hipStream_t stream) {
    const float* x   = (const float*)d_in[0];
    float*       out = (float*)d_out;
    float* ghist = (float*)d_ws;                   // [4][256] f32 accumulator
    float* cdfn  = ghist + NBATCH * N_BINS;        // [4][256] f32 normalized CDF

    const int pixPerBatch = in_sizes[0] / NBATCH;  // 262144

    // 1) zero accumulators (must happen every call: atomics accumulate)
    zero_ws_kernel<<<1, NBATCH * N_BINS, 0, stream>>>(ghist);

    // 2) windowed soft histogram
    const int pixPerBlock = 2048;
    const int blocks1 = (pixPerBatch / pixPerBlock) * NBATCH;   // 512 blocks
    hist_kernel<<<blocks1, 256, 0, stream>>>(x, ghist, pixPerBatch, pixPerBlock);

    // 3) normalize + scan
    cdf_kernel<<<1, N_BINS, 0, stream>>>(ghist, cdfn);

    // 4) WMMA equalization: 8 waves/block, 16 pixels/wave
    const int blocks3 = (pixPerBatch / (16 * 8)) * NBATCH;      // 8192 blocks
    equalize_wmma_kernel<<<blocks3, 256, 0, stream>>>(x, cdfn, out, pixPerBatch);
}
